// MultiheadSpluAttention_33148557590958
// MI455X (gfx1250) — matrix operand order repaired and checked by reading, not compiled, not run on hardware
//
#include <hip/hip_runtime.h>
#include <hip/hip_bf16.h>
#include <math.h>

// ---------------------------------------------------------------------------
// CDNA5 / gfx1250 WMMA types
// ---------------------------------------------------------------------------
typedef __attribute__((ext_vector_type(16))) __bf16 v16bf;
typedef __attribute__((ext_vector_type(8)))  __bf16 v8bf;
typedef __attribute__((ext_vector_type(8)))  float  v8f;

union Frag16 {            // assemble a 16x32 A-fragment from two contiguous 16B runs
  v16bf v;
  v8bf  h[2];
};

#define WMMA_BF16(a, b, c) \
  __builtin_amdgcn_wmma_f32_16x16x32_bf16(false, (a), false, (b), (short)0, (c), false, false)

// Problem constants (from the reference)
constexpr int Lq   = 2048;       // sequence length
constexpr int Bb   = 2;          // batch
constexpr int Ee   = 1024;       // embed dim
constexpr int DQKc = 128;        // per-head q/k dim
constexpr int QKDc = 2048;       // H * DQK
constexpr int DVc  = 64;         // per-head v dim
constexpr int Mrows = Lq * Bb;   // 4096 GEMM rows

// Projection epilogue modes
constexpr int OUT_F32   = 0;  // f32 row-major (final output)
constexpr int OUT_QPAIR = 1;  // bf16: Y0 = head-masked (q1), Y1 = raw (q)
constexpr int OUT_KPAIR = 2;  // bf16: Y0 = head-masked (k1), Y1 = gmask-masked (kg)
constexpr int OUT_VT    = 3;  // bf16 transposed (b, dv, L) for attention B-frags

// ---------------------------------------------------------------------------
// Generic tiled GEMM:  Y = act( X(M,K) * W(N,K)^T + bias ) * scale
// f32 in, bf16 LDS staging, v_wmma_f32_16x16x32_bf16, f32 accumulate.
// Block: 128 threads = 4 waves; block tile 64(M) x 64(N); each wave owns 16x64.
// Fragment loads are wide LDS vector loads (this code's contiguous k = 16*half + s order in both A and B):
//   A: row-major tile, lane reads 2 contiguous 16B runs (ds_load_b128 x2)
//   B: [N][K] tile, lane reads 1 contiguous 32B run
// ALL masking (head masks + global mask) is folded into the epilogue: masks
// depend only on (h,l,d)=(m,n), so applying them here is done once per element
// instead of once per (q-tile, s-tile) pair inside the attention kernel.
// ---------------------------------------------------------------------------
template <bool RELU, int MODE>
__global__ __launch_bounds__(128)
void gemm_bias_kernel(const float* __restrict__ X, const float* __restrict__ W,
                      const float* __restrict__ bias,
                      void* __restrict__ Y0, void* __restrict__ Y1,
                      const unsigned char* __restrict__ mask,
                      const unsigned char* __restrict__ gmaskp,
                      int M, int N, int K, float scale) {
  const int tid  = threadIdx.x;
  const int wave = tid >> 5;
  const int lane = tid & 31;
  const int hlf  = lane >> 4;
  const int idx  = lane & 15;
  const int m0 = blockIdx.x * 64;
  const int n0 = blockIdx.y * 64;

  __shared__ __align__(32) __bf16 sX[64][32];   // rows of X (A source)
  __shared__ __align__(32) __bf16 sW[64][32];   // rows of W (B source, [N][K])

  v8f acc[4] = {};

  for (int k0 = 0; k0 < K; k0 += 32) {
    // Cooperative stage, 8 f32 -> 8 bf16 per step (2 x b128 global, 1 x b128 LDS)
    for (int t = tid; t < (64 * 32) / 8; t += 128) {
      const int r = t >> 2;
      const int c = (t & 3) * 8;
      const float4 x0 = *(const float4*)&X[(size_t)(m0 + r) * K + (k0 + c)];
      const float4 x1 = *(const float4*)&X[(size_t)(m0 + r) * K + (k0 + c) + 4];
      const float4 w0 = *(const float4*)&W[(size_t)(n0 + r) * K + (k0 + c)];
      const float4 w1 = *(const float4*)&W[(size_t)(n0 + r) * K + (k0 + c) + 4];
      v8bf xb, wb;
      xb[0] = (__bf16)x0.x; xb[1] = (__bf16)x0.y; xb[2] = (__bf16)x0.z; xb[3] = (__bf16)x0.w;
      xb[4] = (__bf16)x1.x; xb[5] = (__bf16)x1.y; xb[6] = (__bf16)x1.z; xb[7] = (__bf16)x1.w;
      wb[0] = (__bf16)w0.x; wb[1] = (__bf16)w0.y; wb[2] = (__bf16)w0.z; wb[3] = (__bf16)w0.w;
      wb[4] = (__bf16)w1.x; wb[5] = (__bf16)w1.y; wb[6] = (__bf16)w1.z; wb[7] = (__bf16)w1.w;
      *(v8bf*)&sX[r][c] = xb;
      *(v8bf*)&sW[r][c] = wb;
      if (k0 + 32 < K) {   // gfx1250 global_prefetch_b8 for next K slab
        __builtin_prefetch(&X[(size_t)(m0 + r) * K + (k0 + 32 + c)], 0, 1);
        __builtin_prefetch(&W[(size_t)(n0 + r) * K + (k0 + 32 + c)], 0, 1);
      }
    }
    __syncthreads();

    Frag16 af;
    af.h[0] = *(const v8bf*)&sX[wave * 16 + idx][hlf * 16];
    af.h[1] = *(const v8bf*)&sX[wave * 16 + idx][hlf * 16 + 8];

#pragma unroll
    for (int nb = 0; nb < 4; nb++) {
      const v16bf bfr = *(const v16bf*)&sW[nb * 16 + idx][hlf * 16];
      acc[nb] = WMMA_BF16(af.v, bfr, acc[nb]);
    }
    __syncthreads();
  }

  // Epilogue: bias, relu, scale; C/D layout: M = r + 8*half, N = idx (+16*nb)
#pragma unroll
  for (int nb = 0; nb < 4; nb++) {
#pragma unroll
    for (int r = 0; r < 8; r++) {
      const int m = m0 + wave * 16 + r + (hlf << 3);
      const int n = n0 + nb * 16 + idx;
      float v = acc[nb][r] + bias[n];
      if (RELU) v = fmaxf(v, 0.0f);
      v *= scale;
      if (MODE == OUT_F32) {
        ((float*)Y0)[(size_t)m * N + n] = v;
      } else if (MODE == OUT_QPAIR || MODE == OUT_KPAIR) {
        const __bf16 bv = (__bf16)v;
        const int l = m / Bb, hh = n >> 7, d = n & 127;
        const bool mk = mask[((size_t)hh * Lq + l) * DQKc + d] != 0;
        ((__bf16*)Y0)[(size_t)m * N + n] = mk ? (__bf16)0.0f : bv;  // head-masked
        if (MODE == OUT_QPAIR) {
          ((__bf16*)Y1)[(size_t)m * N + n] = bv;                    // raw q
        } else {
          // gmask rows are uniform over d -> one byte per row decides the row
          const bool gk = gmaskp[(size_t)l * DQKc] != 0;
          ((__bf16*)Y1)[(size_t)m * N + n] = gk ? (__bf16)0.0f : bv; // kg
        }
      } else {  // OUT_VT: (b, dv, L) so attention v B-frags are contiguous
        const int l = m / Bb, bb = m % Bb;
        ((__bf16*)Y0)[((size_t)bb * Ee + n) * Lq + l] = (__bf16)v;
      }
    }
  }
}

// ---------------------------------------------------------------------------
// SPLU attention: per (b, h, 16-row q tile), one wave streams 32-wide s tiles.
// All A/B fragments load directly from global (L2-resident); only the bf16
// score tile goes through LDS to re-layout between the two WMMA stages.
//   interior tiles : scores = q1(16x128) x k1^T        -> 4 WMMA per subtile
//   boundary tiles : + q(16x128) x kg^T (precomputed)  -> +4 WMMA per subtile
//   out += bf16(scores)(16x32) x v(32x64)              -> 4 WMMA
//   den += bf16(scores) x ones                          -> 1 WMMA (row-sum)
//   out /= max(den, eps)  (normalization is linear -> single streaming pass)
// ---------------------------------------------------------------------------
__global__ __launch_bounds__(32)
void splu_attn_kernel(const __bf16* __restrict__ q1b, const __bf16* __restrict__ qb,
                      const __bf16* __restrict__ k1b, const __bf16* __restrict__ kgb,
                      const __bf16* __restrict__ vbT,
                      float* __restrict__ outp) {
  const int qt = blockIdx.x;   // q tile (16 rows)
  const int h  = blockIdx.y;
  const int b  = blockIdx.z;
  const int lane = threadIdx.x;
  const int hlf  = lane >> 4;
  const int idx  = lane & 15;

  __shared__ __align__(32) __bf16 sSb[16][32];   // bf16 score tile relayout

  // ---- A fragments straight from global: afrag[0..3]=q1, afrag[4..7]=q ----
  const size_t qrow = (size_t)((qt * 16 + idx) * Bb + b) * QKDc + (size_t)h * DQKc;
  v16bf afrag[8];
#pragma unroll
  for (int f = 0; f < 4; f++) {
    Frag16 u;
    u.h[0] = *(const v8bf*)(q1b + qrow + f * 32 + hlf * 16);
    u.h[1] = *(const v8bf*)(q1b + qrow + f * 32 + hlf * 16 + 8);
    afrag[f] = u.v;
    u.h[0] = *(const v8bf*)(qb + qrow + f * 32 + hlf * 16);
    u.h[1] = *(const v8bf*)(qb + qrow + f * 32 + hlf * 16 + 8);
    afrag[4 + f] = u.v;
  }

  v16bf ones;
#pragma unroll
  for (int j = 0; j < 16; j++) ones[j] = (__bf16)1.0f;

  v8f acc[4] = {};
  v8f dacc   = {};

  constexpr int NT = Lq / 32;
  for (int st = 0; st < NT; st++) {
    const int s0 = st * 32;
    // gmask rows are uniform and only the first/last DG=16 rows are kept, so
    // only the first and last 32-row s-tile can contribute the q*kg term.
    const bool boundary = (st == 0) || (st == NT - 1);

    // ---- score subtiles: B-frag = contiguous 32B run of a k row ----
#pragma unroll
    for (int sub = 0; sub < 2; sub++) {
      const int s = s0 + sub * 16 + idx;               // this lane's column
      const size_t krow = (size_t)(s * Bb + b) * QKDc + (size_t)h * DQKc;
      v8f c = {};
#pragma unroll
      for (int f = 0; f < 4; f++) {
        const v16bf bfr = *(const v16bf*)(k1b + krow + f * 32 + hlf * 16);
        c = WMMA_BF16(afrag[f], bfr, c);
      }
      if (boundary) {   // kg already masked in the projection epilogue
#pragma unroll
        for (int f = 0; f < 4; f++) {
          const v16bf kf = *(const v16bf*)(kgb + krow + f * 32 + hlf * 16);
          c = WMMA_BF16(afrag[4 + f], kf, c);
        }
      }
      if (st + 1 < NT)   // gfx1250 global_prefetch_b8 for next s-tile k rows
        __builtin_prefetch(k1b + krow + (size_t)32 * Bb * QKDc, 0, 1);
#pragma unroll
      for (int r = 0; r < 8; r++)
        sSb[r + (hlf << 3)][sub * 16 + idx] = (__bf16)c[r];
    }
    __syncthreads();

    // ---- A fragment of bf16 scores (two contiguous 16B runs) ----
    Frag16 pf;
    pf.h[0] = *(const v8bf*)&sSb[idx][hlf * 16];
    pf.h[1] = *(const v8bf*)&sSb[idx][hlf * 16 + 8];

    // out += scores x v (v B-frag contiguous in transposed layout);
    // den += scores x ones (row-sum lands in every column)
#pragma unroll
    for (int nb = 0; nb < 4; nb++) {
      const v16bf vf = *(const v16bf*)(
          vbT + ((size_t)b * Ee + (size_t)h * DVc + nb * 16 + idx) * Lq + s0 + hlf * 16);
      acc[nb] = WMMA_BF16(pf.v, vf, acc[nb]);
    }
    dacc = WMMA_BF16(pf.v, ones, dacc);

    __syncthreads();   // protect sSb before next iteration's stores
  }

  // Epilogue: divide by row-sum; C/D layout: M = r + 8*half, N = idx (+16*nb)
#pragma unroll
  for (int nb = 0; nb < 4; nb++) {
#pragma unroll
    for (int r = 0; r < 8; r++) {
      const int m = r + (hlf << 3);
      const int l = qt * 16 + m;
      const float dd = fmaxf(dacc[r], 1e-12f);
      outp[(size_t)(l * Bb + b) * Ee + (size_t)h * DVc + nb * 16 + idx] = acc[nb][r] / dd;
    }
  }
}

// ---------------------------------------------------------------------------
// Host-side launch. Inputs (setup_inputs order):
// 0 query 1 key 2 value 3 Wq 4 bq 5 Wk 6 bk 7 Wv 8 bv 9 Wo 10 bo 11 masks 12 gmask
// ---------------------------------------------------------------------------
extern "C" void kernel_launch(void* const* d_in, const int* in_sizes, int n_in,
                              void* d_out, int out_size, void* d_ws, size_t ws_size,
                              hipStream_t stream) {
  const float* query = (const float*)d_in[0];
  const float* key_  = (const float*)d_in[1];
  const float* value = (const float*)d_in[2];
  const float* Wq = (const float*)d_in[3];
  const float* bq = (const float*)d_in[4];
  const float* Wk = (const float*)d_in[5];
  const float* bk = (const float*)d_in[6];
  const float* Wv = (const float*)d_in[7];
  const float* bv = (const float*)d_in[8];
  const float* Wo = (const float*)d_in[9];
  const float* bo = (const float*)d_in[10];
  const unsigned char* masks = (const unsigned char*)d_in[11];
  const unsigned char* gmask = (const unsigned char*)d_in[12];

  // Workspace: q1/q/k1/kg (4 x 16MB bf16) + vT (8MB bf16) + attn out (16MB f32)
  char* ws = (char*)d_ws;
  __bf16* q1b = (__bf16*)ws;                          // 4096 x 2048 bf16 (head-masked q)
  __bf16* qb  = q1b + (size_t)Mrows * QKDc;           // 4096 x 2048 bf16 (raw q)
  __bf16* k1b = qb  + (size_t)Mrows * QKDc;           // 4096 x 2048 bf16 (head-masked k)
  __bf16* kgb = k1b + (size_t)Mrows * QKDc;           // 4096 x 2048 bf16 (gmask-masked k)
  __bf16* vbT = kgb + (size_t)Mrows * QKDc;           // (2, 1024, 2048) bf16 transposed
  float*  ao  = (float*)(vbT + (size_t)Bb * Ee * Lq); // 4096 x 1024 f32

  const float scaling = 1.0f / sqrtf((float)QKDc);    // QKD^-0.5

  const dim3 blk(128);
  // q = relu(query Wq^T + bq) * scaling   -> q1 (head-masked) + q (raw)
  gemm_bias_kernel<true, OUT_QPAIR><<<dim3(Mrows / 64, QKDc / 64), blk, 0, stream>>>(
      query, Wq, bq, q1b, qb, masks, nullptr, Mrows, QKDc, Ee, scaling);
  // k = relu(key Wk^T + bk)               -> k1 (head-masked) + kg (gmask-masked)
  gemm_bias_kernel<true, OUT_KPAIR><<<dim3(Mrows / 64, QKDc / 64), blk, 0, stream>>>(
      key_, Wk, bk, k1b, kgb, masks, gmask, Mrows, QKDc, Ee, 1.0f);
  // v = value Wv^T + bv                   -> transposed (b, dv, L)
  gemm_bias_kernel<false, OUT_VT><<<dim3(Mrows / 64, Ee / 64), blk, 0, stream>>>(
      value, Wv, bv, vbT, nullptr, nullptr, nullptr, Mrows, Ee, Ee, 1.0f);
  // masked/global attention with linear normalization
  splu_attn_kernel<<<dim3(Lq / 16, 16, Bb), dim3(32), 0, stream>>>(
      q1b, qb, k1b, kgb, vbT, ao);
  // out = ao Wo^T + bo  (f32 to d_out)
  gemm_bias_kernel<false, OUT_F32><<<dim3(Mrows / 64, Ee / 64), blk, 0, stream>>>(
      ao, Wo, bo, d_out, nullptr, nullptr, nullptr, Mrows, Ee, Ee, 1.0f);
}
